// KNRM_60619168415746
// MI455X (gfx1250) — matrix operand order repaired and checked by reading, not compiled, not run on hardware
//
#include <hip/hip_runtime.h>

typedef __attribute__((ext_vector_type(16))) _Float16 v16h;
typedef __attribute__((ext_vector_type(8)))  float    v8f;

#define LQ     30
#define LD     1024
#define EDIM   256
#define NK     11
#define CHUNK  64
#define NCHUNK (LD / CHUNK)
#define KSTEPS (EDIM / 32)
#define STR    (EDIM + 8)   // LDS row stride in halfs: 528B, 16B aligned, banks spread

union Frag {
    v16h  v;
    uint4 u[2];
};

__global__ __launch_bounds__(256)
void knrm_fused_kernel(const long long* __restrict__ q,
                       const long long* __restrict__ d,
                       const float*     __restrict__ emb,
                       const float*     __restrict__ W,
                       const float*     __restrict__ bias,
                       float*           __restrict__ out)
{
    __shared__ __align__(16) _Float16 qA[32 * STR];       // 32 query rows (2 zero pad)
    __shared__ __align__(16) _Float16 dB[CHUNK * STR];    // 64 doc rows per chunk
    __shared__ float pool[32 * NK];
    __shared__ float phiW[NK];

    const int b    = blockIdx.x;
    const int tid  = threadIdx.x;
    const int wave = tid >> 5;
    const int lane = tid & 31;
    const int lh   = lane >> 4;   // half-wave select
    const int lr   = lane & 15;   // row/col within tile

    for (int i = tid; i < 32 * NK; i += 256) pool[i] = 0.0f;

    // ---- gather + l2-normalize query rows -> f16 LDS (one wave per row) ----
    for (int r = wave; r < 32; r += 8) {
        _Float16* dst = &qA[r * STR + lane * 8];
        if (r < LQ) {
            const long long idx = q[(long long)b * LQ + r];
            const float* src = emb + idx * (long long)EDIM + lane * 8;
            float4 x0 = *(const float4*)(src);
            float4 x1 = *(const float4*)(src + 4);
            float ss = x0.x*x0.x + x0.y*x0.y + x0.z*x0.z + x0.w*x0.w
                     + x1.x*x1.x + x1.y*x1.y + x1.z*x1.z + x1.w*x1.w;
            #pragma unroll
            for (int o = 16; o >= 1; o >>= 1) ss += __shfl_xor(ss, o, 32);
            const float inv = rsqrtf(fmaxf(ss, 1e-12f));
            dst[0] = (_Float16)(x0.x * inv); dst[1] = (_Float16)(x0.y * inv);
            dst[2] = (_Float16)(x0.z * inv); dst[3] = (_Float16)(x0.w * inv);
            dst[4] = (_Float16)(x1.x * inv); dst[5] = (_Float16)(x1.y * inv);
            dst[6] = (_Float16)(x1.z * inv); dst[7] = (_Float16)(x1.w * inv);
        } else {
            #pragma unroll
            for (int j = 0; j < 8; ++j) dst[j] = (_Float16)0.0f;
        }
    }

    // Prefetch chunk 0 doc rows (128B granularity, 2 per thread = full 64KB chunk)
    {
        const int r   = tid >> 2;                 // 0..63
        const int seg = tid & 3;                  // 4 x 256B regions -> 2 x 128B each
        const long long idx0 = d[(long long)b * LD + r];
        const char* base = (const char*)(emb + idx0 * (long long)EDIM) + seg * 256;
        __builtin_prefetch(base,       0, 3);
        __builtin_prefetch(base + 128, 0, 3);
    }
    __syncthreads();

    const int mt = wave & 1;    // m-tile (query rows 0..15 / 16..31)
    const int nt = wave >> 1;   // n-tile (doc cols, 4 tiles of 16 in a 64-chunk)

    // ---- hoist chunk-invariant A fragments (16x32 f16 layout) into VGPRs ----
    Frag afr[KSTEPS];
    #pragma unroll
    for (int kk = 0; kk < KSTEPS; ++kk) {
        const _Float16* pa = &qA[(mt * 16 + lr) * STR + kk * 32 + lh * 8];
        afr[kk].u[0] = *(const uint4*)(pa);
        afr[kk].u[1] = *(const uint4*)(pa + 16);
    }

    // RBF kernel constants (mu ascending; sigma[0]=0.001 pairs with mu=-0.9)
    const float mu_[NK]  = {-0.9f,-0.7f,-0.5f,-0.3f,-0.1f,0.1f,0.3f,0.5f,0.7f,0.9f,1.0f};
    const float i2s_[NK] = {500000.0f,50.0f,50.0f,50.0f,50.0f,50.0f,50.0f,50.0f,50.0f,50.0f,50.0f};

    float acc[8][NK];
    #pragma unroll
    for (int i = 0; i < 8; ++i)
        #pragma unroll
        for (int k = 0; k < NK; ++k) acc[i][k] = 0.0f;

    const int browoff = (nt * 16 + lr) * STR + lh * 8;   // B fragment row base

    for (int ch = 0; ch < NCHUNK; ++ch) {
        // ---- gather + l2-normalize 64 doc rows -> f16 LDS ----
        for (int r = wave; r < CHUNK; r += 8) {
            const long long idx = d[(long long)b * LD + ch * CHUNK + r];
            const float* src = emb + idx * (long long)EDIM + lane * 8;
            float4 x0 = *(const float4*)(src);
            float4 x1 = *(const float4*)(src + 4);
            float ss = x0.x*x0.x + x0.y*x0.y + x0.z*x0.z + x0.w*x0.w
                     + x1.x*x1.x + x1.y*x1.y + x1.z*x1.z + x1.w*x1.w;
            #pragma unroll
            for (int o = 16; o >= 1; o >>= 1) ss += __shfl_xor(ss, o, 32);
            const float inv = rsqrtf(fmaxf(ss, 1e-12f));
            _Float16* dst = &dB[r * STR + lane * 8];
            dst[0] = (_Float16)(x0.x * inv); dst[1] = (_Float16)(x0.y * inv);
            dst[2] = (_Float16)(x0.z * inv); dst[3] = (_Float16)(x0.w * inv);
            dst[4] = (_Float16)(x1.x * inv); dst[5] = (_Float16)(x1.y * inv);
            dst[6] = (_Float16)(x1.z * inv); dst[7] = (_Float16)(x1.w * inv);
        }

        // ---- prefetch next chunk's doc rows into cache (overlaps WMMA below) ----
        if (ch + 1 < NCHUNK) {
            const int r   = tid >> 2;
            const int seg = tid & 3;
            const long long idxn = d[(long long)b * LD + (ch + 1) * CHUNK + r];
            const char* base = (const char*)(emb + idxn * (long long)EDIM) + seg * 256;
            __builtin_prefetch(base,       0, 3);
            __builtin_prefetch(base + 128, 0, 3);
        }
        __syncthreads();

        // ---- sim tile 16x16: 8 chained v_wmma_f32_16x16x32_f16 over K=256,
        //      with B fragments double-buffered in registers so load(k+1)
        //      overlaps wmma(k) ----
        Frag bb[2];
        {
            const _Float16* pb = &dB[browoff];
            bb[0].u[0] = *(const uint4*)(pb);
            bb[0].u[1] = *(const uint4*)(pb + 16);
        }
        v8f c = {};
        #pragma unroll
        for (int kk = 0; kk < KSTEPS; ++kk) {
            if (kk + 1 < KSTEPS) {
                const _Float16* pb = &dB[browoff + (kk + 1) * 32];
                bb[(kk + 1) & 1].u[0] = *(const uint4*)(pb);
                bb[(kk + 1) & 1].u[1] = *(const uint4*)(pb + 16);
            }
            c = __builtin_amdgcn_wmma_f32_16x16x32_f16(
                    false, afr[kk].v, false, bb[kk & 1].v, (short)0, c, false, false);
        }

        // ---- fused RBF kernel pooling straight off the accumulator ----
        #pragma unroll
        for (int i = 0; i < 8; ++i) {
            const float s = c[i];   // sim at row (mt*16 + i + 8*lh), one doc column
            #pragma unroll
            for (int k = 0; k < NK; ++k) {
                const float dd = s - mu_[k];
                acc[i][k] += __expf(-dd * dd * i2s_[k]);
            }
        }
        __syncthreads();   // dB reused next chunk
    }

    // ---- reduce per-lane partials into LDS pool[query_row][kernel] ----
    const int rowbase = mt * 16 + lh * 8;
    #pragma unroll
    for (int i = 0; i < 8; ++i)
        #pragma unroll
        for (int k = 0; k < NK; ++k)
            atomicAdd(&pool[(rowbase + i) * NK + k], acc[i][k]);
    __syncthreads();

    // ---- log pooling over 30 valid query rows, scale by W ----
    if (tid < NK) {
        float s = 0.0f;
        for (int r = 0; r < LQ; ++r)
            s += __logf(fmaxf(pool[r * NK + tid], 1e-10f));
        phiW[tid] = s * 0.01f * W[tid];   // W is [11,1]
    }
    __syncthreads();
    if (tid == 0) {
        float o = bias[0];
        #pragma unroll
        for (int k = 0; k < NK; ++k) o += phiW[k];
        out[b] = o;
    }
}

extern "C" void kernel_launch(void* const* d_in, const int* in_sizes, int n_in,
                              void* d_out, int out_size, void* d_ws, size_t ws_size,
                              hipStream_t stream) {
    const long long* q    = (const long long*)d_in[0];   // [256,30]   int64
    const long long* d    = (const long long*)d_in[1];   // [256,1024] int64
    const float*     emb  = (const float*)d_in[2];       // [200000,256] f32
    const float*     W    = (const float*)d_in[3];       // [11,1]
    const float*     bias = (const float*)d_in[4];       // [1]
    float*           out  = (float*)d_out;               // [256,1]
    (void)in_sizes; (void)n_in; (void)out_size; (void)d_ws; (void)ws_size;

    knrm_fused_kernel<<<dim3(256), dim3(256), 0, stream>>>(q, d, emb, W, bias, out);
}
